// HaarBlock_68908455297113
// MI455X (gfx1250) — 8-bit matrix operand order repaired and checked by reading, compiled, not run on hardware
//
#include <hip/hip_runtime.h>
#include <cstdint>
#include <cstddef>

typedef __attribute__((ext_vector_type(8))) int v8i;

#define CIN   14
#define HMID  14
#define COUT  128
#define BATCH 8
#define HH    384
#define WW    384
#define OH    131
#define OW    131

static constexpr int NX    = BATCH * CIN * HH * WW;
static constexpr int NPIX  = BATCH * OH * OW;
static constexpr int NPART = 1024;
static constexpr int PIX_PER_BLOCK = 64;
static constexpr int OPLANE = OH * OW;

static constexpr size_t WS_PART = 0;
static constexpr size_t WS_MEAN = 4096;
static constexpr size_t WS_W1S  = 4160;
static constexpr size_t WS_B1R  = 4864;
static constexpr size_t WS_W2S  = 4928;
static constexpr size_t WS_B2R  = 6784;
static constexpr size_t WS_W3S  = 7296;
static constexpr size_t WS_B3R  = 23744;
static constexpr size_t WS_XB   = 24576;
static constexpr size_t WS_Y1   = WS_XB + (size_t)NX;

__global__ void k_reduce(const float* __restrict__ x, float* __restrict__ partials, int n) {
    __shared__ float sm[256];
    const int stride = gridDim.x * blockDim.x;
    float s = 0.f;
    for (int i = blockIdx.x * blockDim.x + threadIdx.x; i < n; i += stride) {
        __builtin_prefetch(x + i + stride, 0, 0);
        s += x[i];
    }
    sm[threadIdx.x] = s;
    __syncthreads();
    for (int off = 128; off > 0; off >>= 1) {
        if ((int)threadIdx.x < off) sm[threadIdx.x] += sm[threadIdx.x + off];
        __syncthreads();
    }
    if (threadIdx.x == 0) partials[blockIdx.x] = sm[0];
}

__global__ void k_finalize(const float* __restrict__ partials, float* __restrict__ meanp,
                           int nPart, float invN) {
    __shared__ float sm[256];
    float s = 0.f;
    for (int i = threadIdx.x; i < nPart; i += 256) s += partials[i];
    sm[threadIdx.x] = s;
    __syncthreads();
    for (int off = 128; off > 0; off >>= 1) {
        if ((int)threadIdx.x < off) sm[threadIdx.x] += sm[threadIdx.x + off];
        __syncthreads();
    }
    if (threadIdx.x == 0) *meanp = sm[0] * invN;
}

__global__ void k_prep(const float* w1, const float* b1, const float* w2, const float* b2,
                       const float* w3, const float* b3,
                       int8_t* w1s, int* b1r, int8_t* w2s, int* b2r, int8_t* w3s, int* b3r) {
    int t = threadIdx.x;
    for (int i = t; i < HMID * 49;   i += 256) w1s[i] = (w1[i] >= 0.f) ? 1 : -1;
    for (int i = t; i < HMID;        i += 256) b1r[i] = (int)rintf(b1[i]);
    for (int i = t; i < COUT * CIN;  i += 256) w2s[i] = (w2[i] >= 0.f) ? 1 : -1;
    for (int i = t; i < COUT;        i += 256) b2r[i] = (int)rintf(b2[i]);
    for (int i = t; i < COUT * COUT; i += 256) w3s[i] = (w3[i] >= 0.f) ? 1 : -1;
    for (int i = t; i < COUT;        i += 256) b3r[i] = (int)rintf(b3[i]);
}

__global__ void k_binarize(const float* __restrict__ x, const float* __restrict__ meanp,
                           int8_t* __restrict__ xb, int n) {
    float m = *meanp;
    for (int i = blockIdx.x * blockDim.x + threadIdx.x; i < n; i += gridDim.x * blockDim.x)
        xb[i] = (x[i] - m) >= 0.f ? (int8_t)1 : (int8_t)-1;
}

__global__ void k_conv1(const int8_t* __restrict__ xb, const int8_t* __restrict__ w1s,
                        const int* __restrict__ b1r, int8_t* __restrict__ y1) {
    int i = blockIdx.x * blockDim.x + threadIdx.x;
    if (i >= NX / 4) return;
    int xq  = i % (WW / 4);   int t = i / (WW / 4);
    int xw4 = 4 * xq;
    int yh  = t % HH;  t /= HH;
    int m   = t % CIN; int b = t / CIN;
    const int8_t* src  = xb + ((size_t)(b * CIN + m) * HH) * WW;
    const int8_t* wrow = w1s + m * 49;
    int acc0 = 0, acc1 = 0, acc2 = 0, acc3 = 0;
    #pragma unroll
    for (int dy = 0; dy < 7; ++dy) {
        int iy = yh + dy - 3;
        if ((unsigned)iy >= (unsigned)HH) continue;
        const int8_t* row = src + iy * WW;
        __builtin_prefetch(row + xw4 + WW, 0, 0);
        int v[10];
        #pragma unroll
        for (int k = 0; k < 10; ++k) {
            int ix = xw4 - 3 + k;
            v[k] = ((unsigned)ix < (unsigned)WW) ? (int)row[ix] : 0;
        }
        #pragma unroll
        for (int dx = 0; dx < 7; ++dx) {
            int wv = (int)wrow[dy * 7 + dx];
            acc0 += v[dx + 0] * wv;
            acc1 += v[dx + 1] * wv;
            acc2 += v[dx + 2] * wv;
            acc3 += v[dx + 3] * wv;
        }
    }
    int bias = b1r[m];
    size_t o = ((size_t)(b * CIN + m) * HH + yh) * WW + xw4;
    y1[o + 0] = (acc0 + bias) >= 0 ? (int8_t)1 : (int8_t)-1;
    y1[o + 1] = (acc1 + bias) >= 0 ? (int8_t)1 : (int8_t)-1;
    y1[o + 2] = (acc2 + bias) >= 0 ? (int8_t)1 : (int8_t)-1;
    y1[o + 3] = (acc3 + bias) >= 0 ? (int8_t)1 : (int8_t)-1;
}

__global__ void __launch_bounds__(256)
k_conv23(const int8_t* __restrict__ y1, const int8_t* __restrict__ w2s,
         const int* __restrict__ b2r, const int8_t* __restrict__ w3s,
         const int* __restrict__ b3r, float* __restrict__ out) {
    __shared__ __align__(16) int8_t sB[PIX_PER_BLOCK][128];
    __shared__ int sOBase[PIX_PER_BLOCK];

    const int tile = blockIdx.x;
    const int t    = threadIdx.x;

    {
        int p  = t & 63;
        int cg = t >> 6;
        int g  = tile * PIX_PER_BLOCK + p;
        bool zero = true, inb = false;
        int b = 0, iy = 0, ix = 0, obase = -1;
        if (g < NPIX) {
            int ox = g % OW; int r = g / OW;
            int oy = r % OH; b = r / OH;
            obase = b * (COUT * OPLANE) + oy * OW + ox;
            int iy390 = 3 * oy - 1, ix390 = 3 * ox - 1;
            if (iy390 >= 0 && ix390 >= 0) {
                zero = false;
                iy = iy390 - 3; ix = ix390 - 3;
                inb = (unsigned)iy < (unsigned)HH && (unsigned)ix < (unsigned)WW;
            }
        }
        if (cg == 0) sOBase[p] = obase;
        int yv[CIN];
        if (inb) {
            #pragma unroll
            for (int k = 0; k < CIN; ++k)
                yv[k] = (int)y1[((size_t)(b * CIN + k) * HH + iy) * WW + ix];
        }
        #pragma unroll 4
        for (int j = 0; j < 32; ++j) {
            int c = cg * 32 + j;
            int8_t v;
            if (zero) {
                v = 0;
            } else {
                int acc = b2r[c];
                if (inb) {
                    const int8_t* wr = w2s + c * CIN;
                    #pragma unroll
                    for (int k = 0; k < CIN; ++k) acc += yv[k] * (int)wr[k];
                }
                v = acc >= 0 ? (int8_t)1 : (int8_t)-1;
            }
            sB[p][c] = v;
        }
    }
    __syncthreads();

    const int wave = t >> 5;
    const int lane = t & 31;
    const int hi   = lane >> 4;
    const int lo   = lane & 15;

    v8i a0, a1;
    {
        const int8_t* arow = w3s + (size_t)(16 * wave + lo) * 128;
        #pragma unroll
        for (int j = 0; j < 8; ++j) {
            int k0 = 16 * (j >> 1) + 4 * (j & 1) + 8 * hi;
            a0[j] = *(const int*)(arow + k0);
            a1[j] = *(const int*)(arow + k0 + 64);
        }
    }
    int thr[8];
    #pragma unroll
    for (int rr = 0; rr < 8; ++rr) thr[rr] = -b3r[16 * wave + 8 * hi + rr];

    float* outc = out + (size_t)(16 * wave + 8 * hi) * OPLANE;

    #pragma unroll
    for (int nt = 0; nt < PIX_PER_BLOCK / 16; ++nt) {
        v8i b0, b1;
        const int8_t* bcol = &sB[nt * 16 + lo][0];
        #pragma unroll
        for (int j = 0; j < 8; ++j) {
            int k0 = 16 * (j >> 1) + 4 * (j & 1) + 8 * hi;
            b0[j] = *(const int*)(bcol + k0);
            b1[j] = *(const int*)(bcol + k0 + 64);
        }
        v8i acc = {};
        acc = __builtin_amdgcn_wmma_i32_16x16x64_iu8(true, a0, true, b0, acc, false, false);
        acc = __builtin_amdgcn_wmma_i32_16x16x64_iu8(true, a1, true, b1, acc, false, false);

        int ob = sOBase[nt * 16 + lo];
        if (ob >= 0) {
            #pragma unroll
            for (int rr = 0; rr < 8; ++rr)
                outc[(size_t)ob + (size_t)rr * OPLANE] = (acc[rr] >= thr[rr]) ? 1.0f : -1.0f;
        }
    }
}

extern "C" void kernel_launch(void* const* d_in, const int* in_sizes, int n_in,
                              void* d_out, int out_size, void* d_ws, size_t ws_size,
                              hipStream_t stream) {
    const float* x  = (const float*)d_in[0];
    const float* w1 = (const float*)d_in[1];
    const float* b1 = (const float*)d_in[2];
    const float* w2 = (const float*)d_in[3];
    const float* b2 = (const float*)d_in[4];
    const float* w3 = (const float*)d_in[5];
    const float* b3 = (const float*)d_in[6];
    float* out = (float*)d_out;
    char*  ws  = (char*)d_ws;

    float*  partials = (float*)(ws + WS_PART);
    float*  meanp    = (float*)(ws + WS_MEAN);
    int8_t* w1s      = (int8_t*)(ws + WS_W1S);
    int*    b1r      = (int*)(ws + WS_B1R);
    int8_t* w2s      = (int8_t*)(ws + WS_W2S);
    int*    b2r      = (int*)(ws + WS_B2R);
    int8_t* w3s      = (int8_t*)(ws + WS_W3S);
    int*    b3r      = (int*)(ws + WS_B3R);
    int8_t* xb       = (int8_t*)(ws + WS_XB);
    int8_t* y1       = (int8_t*)(ws + WS_Y1);

    k_reduce  <<<NPART, 256, 0, stream>>>(x, partials, NX);
    k_finalize<<<1, 256, 0, stream>>>(partials, meanp, NPART, 1.0f / (float)NX);
    k_prep    <<<1, 256, 0, stream>>>(w1, b1, w2, b2, w3, b3, w1s, b1r, w2s, b2r, w3s, b3r);
    k_binarize<<<(NX + 255) / 256, 256, 0, stream>>>(x, meanp, xb, NX);

    int c1threads = NX / 4;
    k_conv1   <<<(c1threads + 255) / 256, 256, 0, stream>>>(xb, w1s, b1r, y1);

    int tiles = (NPIX + PIX_PER_BLOCK - 1) / PIX_PER_BLOCK;
    k_conv23  <<<tiles, 256, 0, stream>>>(y1, w2s, b2r, w3s, b3r, out);
}
